// SpatialTemporalAttention_34969623724527
// MI455X (gfx1250) — matrix operand order repaired and checked by reading, not compiled, not run on hardware
//
#include <hip/hip_runtime.h>
#include <hip/hip_bf16.h>
#include <stdint.h>

// Problem constants (B, C, D, H, W) = (2, 256, 16, 64, 64)
#define BB 2
#define CC 256
#define DD 16
#define HH 64
#define WW 64
#define SS (DD * HH * WW)                  // 65536 spatial points per batch
#define ACT_ELEMS ((size_t)BB * CC * SS)   // 33,554,432 elements per activation

typedef __bf16 bf16_t;
typedef __attribute__((ext_vector_type(16))) __bf16 v16bf;
typedef __attribute__((ext_vector_type(8)))  __bf16 bf16x8;
typedef __attribute__((ext_vector_type(8)))  float  v8f;
typedef __attribute__((ext_vector_type(4)))  float  v4f;

// ---------------------------------------------------------------------------
// CDNA5 async global->LDS copy (ASYNCcnt path), per CDNA5 ISA
// GVS mode: saddr = uniform 64-bit base, vaddr = per-lane 32-bit byte offset,
// first VGPR operand = LDS byte address.
// ---------------------------------------------------------------------------
__device__ __forceinline__ void async_copy_b128(uint32_t lds_byte_off,
                                                uint64_t gbase,
                                                uint32_t gbyte_off) {
  asm volatile("global_load_async_to_lds_b128 %0, %1, %2"
               :: "v"(lds_byte_off), "v"(gbyte_off), "s"(gbase)
               : "memory");
}
__device__ __forceinline__ void wait_async0() {
  asm volatile("s_wait_asynccnt 0" ::: "memory");
}

__device__ __forceinline__ uint32_t lds_off_of(const void* p) {
  // Flat LDS addresses keep the LDS byte offset in the low 32 bits (ISA 10.2).
  return (uint32_t)(uintptr_t)p;
}

__device__ __forceinline__ v8f wmma_bf16(v16bf a, v16bf b, v8f c) {
  // D = A(16x32 bf16) * B(32x16 bf16) + C(16x16 f32)
  return __builtin_amdgcn_wmma_f32_16x16x32_bf16(false, a, false, b,
                                                 (short)0, c, false, false);
}

union FragU { v16bf v; bf16x8 h[2]; };

// ---------------------------------------------------------------------------
// Weight conversion: f32 [256x256] -> bf16
// ---------------------------------------------------------------------------
__global__ __launch_bounds__(256) void cvt_w_kernel(const float* __restrict__ w,
                                                    bf16_t* __restrict__ o) {
  int i = blockIdx.x * 256 + threadIdx.x;   // grid covers 65536
  o[i] = (bf16_t)w[i];
}

// ---------------------------------------------------------------------------
// 1x1x1 conv == GEMM:  Out[o,s] = act( sum_c W[o,c] * X[c,s] + bias[o] )
// Block: 256 threads (8 waves). Workgroup tile: 64 (o) x 256 (s). K = 256.
// Wave tile: 32 x 64 -> 2 A-frags x 4 B-frags = 8 accumulators, 8 WMMA/K-step.
// Double-buffered LDS: async weight copy + global X gather for step k+1
// overlap the WMMAs of step k; one barrier per K-step.
// grid = (SS/256, CC/64, BB)
// ---------------------------------------------------------------------------
template <bool IN_F32, bool RELU, bool OUT_F32>
__global__ __launch_bounds__(256)
void conv1x1_kernel(const void* __restrict__ xin,      // [B][C][S] f32 or bf16
                    const bf16_t* __restrict__ wbf,    // [256][256] bf16
                    const float* __restrict__ bias,    // [256] f32
                    void* __restrict__ out) {          // [B][C][S] bf16 or f32
  __shared__ bf16_t Ash[2][64][40];    // weight tile [m][k], rows padded to 80 B
  __shared__ bf16_t Xsh[2][256][40];   // activation tile transposed [n][k]

  const int tid  = threadIdx.x;
  const int lane = tid & 31;
  const int wave = tid >> 5;     // 0..7
  const int wm   = wave >> 2;    // 0..1 : 32-row M group
  const int wn   = wave & 3;     // 0..3 : 64-col N group
  const int ml   = lane & 15;
  const int half = lane >> 4;

  const int b  = blockIdx.z;
  const int o0 = blockIdx.y * 64;
  const int n0 = blockIdx.x * 256;
  const size_t act_base = (size_t)b * CC * SS;

  const uint32_t ash_base = lds_off_of(&Ash[0][0][0]);
  const uint32_t ash_bufb = 64 * 40 * 2;   // bytes per A buffer

  // ---- stagers -------------------------------------------------------------
  auto stageA = [&](int k0, int buf) {
    // 64 rows x 64 bytes, 4 x b128 chunks/row, one chunk per thread (async)
    const int r = tid >> 2;             // row 0..63
    const int q = tid & 3;              // chunk 0..3
    const uint32_t lds = ash_base + (uint32_t)buf * ash_bufb +
                         (uint32_t)(r * 40 + q * 8) * 2u;
    const uint32_t gof = (uint32_t)(((o0 + r) * CC + k0) * 2 + q * 16);
    async_copy_b128(lds, (uint64_t)(uintptr_t)wbf, gof);
  };

  auto stageX = [&](int k0, int buf) {
    // Thread owns column n = tid; gathers 32 k values as 4 x b128 LDS stores.
    // Global side: for fixed k, lanes read consecutive n -> coalesced.
    const int n = tid;
#pragma unroll
    for (int g = 0; g < 4; ++g) {
      bf16x8 tmp;
      if (IN_F32) {
        const float* X = (const float*)xin + act_base + n0 + n;
#pragma unroll
        for (int j = 0; j < 8; ++j)
          tmp[j] = (bf16_t)X[(size_t)(k0 + g * 8 + j) * SS];
      } else {
        const bf16_t* X = (const bf16_t*)xin + act_base + n0 + n;
#pragma unroll
        for (int j = 0; j < 8; ++j)
          tmp[j] = X[(size_t)(k0 + g * 8 + j) * SS];
      }
      *(bf16x8*)&Xsh[buf][n][g * 8] = tmp;
    }
  };

  v8f acc[2][4] = {};

  // ---- prologue ------------------------------------------------------------
  stageA(0, 0);
  stageX(0, 0);
  wait_async0();
  __syncthreads();

  // ---- main K loop (8 steps of 32, double buffered, fully unrolled) -------
#pragma unroll
  for (int ks = 0; ks < 8; ++ks) {
    const int buf = ks & 1;
    if (ks < 7) {                       // prefetch next tiles into other buffer
      stageA((ks + 1) * 32, buf ^ 1);
      stageX((ks + 1) * 32, buf ^ 1);
    }

    // A fragments: rows 32*wm + 16*i + ml, K runs {half*8..+7},{16+half*8..+7}
    FragU af[2];
#pragma unroll
    for (int i = 0; i < 2; ++i) {
      const bf16_t* arow = &Ash[buf][32 * wm + 16 * i + ml][0];
      af[i].h[0] = *(const bf16x8*)(arow + half * 8);
      af[i].h[1] = *(const bf16x8*)(arow + 16 + half * 8);
    }
#pragma unroll
    for (int j = 0; j < 4; ++j) {
      // B fragment: column n, K runs {half*8..+7},{16+half*8..+7}
      FragU bfr;
      const bf16_t* brow = &Xsh[buf][64 * wn + 16 * j + ml][0];
      bfr.h[0] = *(const bf16x8*)(brow + half * 8);
      bfr.h[1] = *(const bf16x8*)(brow + 16 + half * 8);
#pragma unroll
      for (int i = 0; i < 2; ++i)
        acc[i][j] = wmma_bf16(af[i].v, bfr.v, acc[i][j]);
    }

    if (ks < 7) {
      wait_async0();                    // next A tile landed
      __syncthreads();                  // everyone done reading/writing
    }
  }

  // ---- epilogue: bias, optional relu, store (C/D layout: M = r + half*8) ---
#pragma unroll
  for (int i = 0; i < 2; ++i) {
#pragma unroll
    for (int j = 0; j < 4; ++j) {
      const int ng = n0 + 64 * wn + 16 * j + ml;
#pragma unroll
      for (int r = 0; r < 8; ++r) {
        const int o = o0 + 32 * wm + 16 * i + r + half * 8;
        float val = acc[i][j][r] + bias[o];
        if (RELU) val = fmaxf(val, 0.0f);
        const size_t idx = act_base + (size_t)o * SS + ng;
        if (OUT_F32) ((float*)out)[idx] = val;
        else         ((bf16_t*)out)[idx] = (bf16_t)val;
      }
    }
  }
}

// ---------------------------------------------------------------------------
// Attention: one block per (b,c,d) slice (grid = 8192).
//   scores[h,x] = softmax_x( sum_w Q[h,w]*K[x,w] )
//   wv[h,w]     = sum_x scores[h,x] * V[x,w]
// ---------------------------------------------------------------------------
__global__ __launch_bounds__(256)
void attention_kernel(const bf16_t* __restrict__ q,
                      const bf16_t* __restrict__ k,
                      const bf16_t* __restrict__ v,
                      bf16_t* __restrict__ wv) {
  __shared__ bf16_t Qs[64][64];    // [h][w]
  __shared__ bf16_t Ks[64][64];    // [x][w]
  __shared__ bf16_t Vt[64][72];    // transposed [w][x] (padded)
  __shared__ float  Sc[64][64];    // f32 scores
  __shared__ bf16_t Sb[64][64];    // bf16 softmaxed scores

  const int tid  = threadIdx.x;
  const int lane = tid & 31;
  const int wave = tid >> 5;
  const int ml   = lane & 15;
  const int half = lane >> 4;

  const size_t base = (size_t)blockIdx.x * (HH * WW);   // slice of 4096 elems
  const uint32_t qs_base = lds_off_of(&Qs[0][0]);
  const uint32_t ks_base = lds_off_of(&Ks[0][0]);

  // async-stage Q and K (8 KB each = 512 x b128 chunks)
  for (int t = tid; t < 512; t += 256) {
    async_copy_b128(qs_base + t * 16, (uint64_t)(uintptr_t)(q + base), t * 16);
    async_copy_b128(ks_base + t * 16, (uint64_t)(uintptr_t)(k + base), t * 16);
  }
  // stage V transposed (VGPR path)
  for (int idx = tid; idx < HH * WW; idx += 256) {
    const int x = idx >> 6, w = idx & 63;
    Vt[w][x] = v[base + idx];
  }
  wait_async0();
  __syncthreads();

  // scores = Q * K^T  (16 tiles of 16x16; 2 per wave; K-dim = 64)
#pragma unroll
  for (int it = 0; it < 2; ++it) {
    const int t  = wave + it * 8;
    const int mt = (t >> 2) * 16;
    const int nt = (t & 3) * 16;
    v8f acc = {};
#pragma unroll
    for (int k0 = 0; k0 < 64; k0 += 32) {
      FragU af, bfr;
      const bf16_t* arow = &Qs[mt + ml][k0];
      af.h[0] = *(const bf16x8*)(arow + half * 8);
      af.h[1] = *(const bf16x8*)(arow + 16 + half * 8);
      const bf16_t* brow = &Ks[nt + ml][k0];
      bfr.h[0] = *(const bf16x8*)(brow + half * 8);
      bfr.h[1] = *(const bf16x8*)(brow + 16 + half * 8);
      acc = wmma_bf16(af.v, bfr.v, acc);
    }
#pragma unroll
    for (int r = 0; r < 8; ++r) Sc[mt + r + half * 8][nt + ml] = acc[r];
  }
  __syncthreads();

  // row softmax over x (f32)
  if (tid < 64) {
    float mx = -3.4e38f;
#pragma unroll 8
    for (int j = 0; j < 64; ++j) mx = fmaxf(mx, Sc[tid][j]);
    float sum = 0.0f;
#pragma unroll 8
    for (int j = 0; j < 64; ++j) {
      float e = __expf(Sc[tid][j] - mx);
      Sc[tid][j] = e;
      sum += e;
    }
    const float inv = 1.0f / sum;
#pragma unroll 8
    for (int j = 0; j < 64; ++j) Sb[tid][j] = (bf16_t)(Sc[tid][j] * inv);
  }
  __syncthreads();

  // wv = scores * V  (A = Sb [h][x], B via Vt[w][x] -> B[k=x][n=w])
#pragma unroll
  for (int it = 0; it < 2; ++it) {
    const int t  = wave + it * 8;
    const int mt = (t >> 2) * 16;
    const int nt = (t & 3) * 16;
    v8f acc = {};
#pragma unroll
    for (int k0 = 0; k0 < 64; k0 += 32) {
      FragU af, bfr;
      const bf16_t* arow = &Sb[mt + ml][k0];
      af.h[0] = *(const bf16x8*)(arow + half * 8);
      af.h[1] = *(const bf16x8*)(arow + 16 + half * 8);
      const bf16_t* brow = &Vt[nt + ml][k0];
      bfr.h[0] = *(const bf16x8*)(brow + half * 8);
      bfr.h[1] = *(const bf16x8*)(brow + 16 + half * 8);
      acc = wmma_bf16(af.v, bfr.v, acc);
    }
#pragma unroll
    for (int r = 0; r < 8; ++r)
      wv[base + (size_t)(mt + r + half * 8) * WW + nt + ml] = (bf16_t)acc[r];
  }
}

// ---------------------------------------------------------------------------
// Host-side orchestration.
// d_in order: eeg, image, q_w, q_b, k_w, k_b, v_w, v_b, attn_w, attn_b, out_w, out_b
// Workspace layout (bytes):
//   [0, 640K)        : 5 bf16 weight matrices (q,k,v,attn,out)
//   [1M, 1M+64M)     : q  (bf16)  -- reused for 'a' after attention
//   [+64M)           : k  (bf16)
//   [+128M)          : v  (bf16)
//   [+192M)          : wv (bf16)
// ---------------------------------------------------------------------------
extern "C" void kernel_launch(void* const* d_in, const int* in_sizes, int n_in,
                              void* d_out, int out_size, void* d_ws, size_t ws_size,
                              hipStream_t stream) {
  (void)in_sizes; (void)n_in; (void)out_size; (void)ws_size;

  const float* eeg  = (const float*)d_in[0];
  const float* img  = (const float*)d_in[1];
  const float* q_w  = (const float*)d_in[2];
  const float* q_b  = (const float*)d_in[3];
  const float* k_w  = (const float*)d_in[4];
  const float* k_b  = (const float*)d_in[5];
  const float* v_w  = (const float*)d_in[6];
  const float* v_b  = (const float*)d_in[7];
  const float* a_w  = (const float*)d_in[8];
  const float* a_b  = (const float*)d_in[9];
  const float* o_w  = (const float*)d_in[10];
  const float* o_b  = (const float*)d_in[11];

  char* ws = (char*)d_ws;
  bf16_t* wq = (bf16_t*)ws;
  bf16_t* wk = wq + 65536;
  bf16_t* wv = wk + 65536;
  bf16_t* wa = wv + 65536;
  bf16_t* wo = wa + 65536;

  const size_t ACTB = ACT_ELEMS * sizeof(bf16_t);     // 64 MiB
  bf16_t* qbuf  = (bf16_t*)(ws + (1u << 20));
  bf16_t* kbuf  = (bf16_t*)((char*)qbuf + ACTB);
  bf16_t* vbuf  = (bf16_t*)((char*)kbuf + ACTB);
  bf16_t* wvbuf = (bf16_t*)((char*)vbuf + ACTB);
  bf16_t* abuf  = qbuf;   // q dead after attention

  // weight conversions
  cvt_w_kernel<<<256, 256, 0, stream>>>(q_w, wq);
  cvt_w_kernel<<<256, 256, 0, stream>>>(k_w, wk);
  cvt_w_kernel<<<256, 256, 0, stream>>>(v_w, wv);
  cvt_w_kernel<<<256, 256, 0, stream>>>(a_w, wa);
  cvt_w_kernel<<<256, 256, 0, stream>>>(o_w, wo);

  const dim3 gconv(SS / 256, CC / 64, BB);   // (256, 4, 2)
  // q/k/v projections (f32 in, bf16 out, no relu)
  conv1x1_kernel<true, false, false><<<gconv, 256, 0, stream>>>(eeg, wq, q_b, qbuf);
  conv1x1_kernel<true, false, false><<<gconv, 256, 0, stream>>>(img, wk, k_b, kbuf);
  conv1x1_kernel<true, false, false><<<gconv, 256, 0, stream>>>(img, wv, v_b, vbuf);

  // attention over 8192 (b,c,d) slices
  attention_kernel<<<BB * CC * DD, 256, 0, stream>>>(qbuf, kbuf, vbuf, wvbuf);

  // attn conv + relu (bf16 in, bf16 out), then out conv + relu (f32 out)
  conv1x1_kernel<false, true, false><<<gconv, 256, 0, stream>>>(wvbuf, wa, a_b, abuf);
  conv1x1_kernel<false, true, true ><<<gconv, 256, 0, stream>>>(abuf, wo, o_b, (float*)d_out);
}
